// MultiHeadAttentionWithBias_9328668966949
// MI455X (gfx1250) — matrix operand order repaired and checked by reading, not compiled, not run on hardware
//
#include <hip/hip_runtime.h>
#include <stdint.h>

typedef _Float16 v16h __attribute__((ext_vector_type(16)));
typedef _Float16 v8h  __attribute__((ext_vector_type(8)));
typedef float    v8f  __attribute__((ext_vector_type(8)));

#define D_MODEL 512
#define N_HEADS 8
#define HEAD_DIM 64
#define SEQ 2048
#define BATCH 2
#define TOPK 64
#define MROWS (BATCH * SEQ) /* 4096 */
#define SCALE 0.125f        /* 64^-0.5 */

#define WMMA_F16(A, B, C) \
  __builtin_amdgcn_wmma_f32_16x16x32_f16(false, (A), false, (B), (short)0, (C), false, false)

__device__ __forceinline__ v16h cat8(v8h lo, v8h hi) {
  return __builtin_shufflevector(lo, hi, 0, 1, 2, 3, 4, 5, 6, 7, 8, 9, 10, 11, 12, 13, 14, 15);
}

// A-fragment (16x32 f16, MxK): lane = row m; halves 0..15 = K[k0 + lh*16 .. +16)
// contiguous (lh = lane>>4), matching the B-fragment k order.
__device__ __forceinline__ v16h afrag_from_f32(const float* __restrict__ p, int k0, int lh) {
  const float4* a = reinterpret_cast<const float4*>(p + k0 + lh * 16);
  const float4* b = reinterpret_cast<const float4*>(p + k0 + lh * 16 + 8);
  float4 x0 = a[0], x1 = a[1], y0 = b[0], y1 = b[1];
  v16h r;
  r[0] = (_Float16)x0.x;  r[1] = (_Float16)x0.y;  r[2] = (_Float16)x0.z;  r[3] = (_Float16)x0.w;
  r[4] = (_Float16)x1.x;  r[5] = (_Float16)x1.y;  r[6] = (_Float16)x1.z;  r[7] = (_Float16)x1.w;
  r[8] = (_Float16)y0.x;  r[9] = (_Float16)y0.y;  r[10] = (_Float16)y0.z; r[11] = (_Float16)y0.w;
  r[12] = (_Float16)y1.x; r[13] = (_Float16)y1.y; r[14] = (_Float16)y1.z; r[15] = (_Float16)y1.w;
  return r;
}

__device__ __forceinline__ v16h afrag_from_f16(const _Float16* __restrict__ p, int k0, int lh) {
  v8h lo = *reinterpret_cast<const v8h*>(p + k0 + lh * 16);
  v8h hi = *reinterpret_cast<const v8h*>(p + k0 + lh * 16 + 8);
  return cat8(lo, hi);
}

// B-fragment (32x16 f16, KxN): lane holds column n = lane&15,
// K-range k0 + (lane>>4)*16 .. +16 contiguous.
__device__ __forceinline__ v16h bfrag_from_f32(const float* __restrict__ p, int k0, int lh) {
  const float4* a = reinterpret_cast<const float4*>(p + k0 + lh * 16);
  float4 x0 = a[0], x1 = a[1], x2 = a[2], x3 = a[3];
  v16h r;
  r[0] = (_Float16)x0.x;  r[1] = (_Float16)x0.y;  r[2] = (_Float16)x0.z;  r[3] = (_Float16)x0.w;
  r[4] = (_Float16)x1.x;  r[5] = (_Float16)x1.y;  r[6] = (_Float16)x1.z;  r[7] = (_Float16)x1.w;
  r[8] = (_Float16)x2.x;  r[9] = (_Float16)x2.y;  r[10] = (_Float16)x2.z; r[11] = (_Float16)x2.w;
  r[12] = (_Float16)x3.x; r[13] = (_Float16)x3.y; r[14] = (_Float16)x3.z; r[15] = (_Float16)x3.w;
  return r;
}

__device__ __forceinline__ v16h bfrag_from_f16(const _Float16* __restrict__ p, int k0, int lh) {
  return *reinterpret_cast<const v16h*>(p + k0 + lh * 16);
}

__device__ __forceinline__ float ord2f(uint32_t u) {
  uint32_t bits = (u & 0x80000000u) ? (u ^ 0x80000000u) : ~u;
  return __uint_as_float(bits);
}

// ---------------------------------------------------------------------------
// GEMM: C[M x 512] = X[M x 512] @ W^T + bias.  256 thr = 8 waves; block tile
// 128(M) x 64(N); each wave: 16 rows x 64 cols = 4 WMMA accumulators,
// K swept in 16 steps of 32 -> 64 v_wmma per wave.
// MODE 0: store f16 head-major [B][H][N][64].  MODE 1: store fp32 row-major.
// ---------------------------------------------------------------------------
template <typename XT, int MODE>
__global__ __launch_bounds__(256) void proj_gemm_kernel(const XT* __restrict__ X,
                                                        const float* __restrict__ W,
                                                        const float* __restrict__ bias,
                                                        void* __restrict__ outp) {
  const int lane = threadIdx.x & 31;
  const int wave = threadIdx.x >> 5;
  const int lrow = lane & 15;
  const int lh = lane >> 4;
  const int m0 = blockIdx.y * 128 + wave * 16;
  const int n0 = blockIdx.x * 64;

  const XT* xr = X + (size_t)(m0 + lrow) * D_MODEL;
  v8f acc[4] = {};

  for (int k0 = 0; k0 < D_MODEL; k0 += 32) {
    v16h a;
    if constexpr (sizeof(XT) == 4)
      a = afrag_from_f32((const float*)xr, k0, lh);
    else
      a = afrag_from_f16((const _Float16*)xr, k0, lh);
#pragma unroll
    for (int tt = 0; tt < 4; ++tt) {
      int j = n0 + tt * 16 + lrow;
      v16h b = bfrag_from_f32(W + (size_t)j * D_MODEL, k0, lh);
      acc[tt] = WMMA_F16(a, b, acc[tt]);
    }
  }

#pragma unroll
  for (int tt = 0; tt < 4; ++tt) {
    const int j = n0 + tt * 16 + lrow;
    const float bj = bias[j];
#pragma unroll
    for (int r = 0; r < 8; ++r) {
      const int m = m0 + lh * 8 + r;  // C layout: lanes 16-31 hold M = 8 + r
      const float v = acc[tt][r] + bj;
      if constexpr (MODE == 0) {
        const int b = m >> 11, nseq = m & (SEQ - 1);
        const int h = j >> 6, d = j & 63;
        ((_Float16*)outp)[(((size_t)(b * N_HEADS + h) * SEQ + nseq) << 6) + d] = (_Float16)v;
      } else {
        ((float*)outp)[(size_t)m * D_MODEL + j] = v;
      }
    }
  }
}

// ---------------------------------------------------------------------------
// Scores: S[bh, q, k] = scale * Q . K.  Block = 16 q-rows x 2048 keys;
// each of 8 waves covers 256 keys -> 16 key-tiles x 2 wmma (d = 2 chains of 32).
// ---------------------------------------------------------------------------
__global__ __launch_bounds__(256) void attn_scores_kernel(const _Float16* __restrict__ Qhm,
                                                          const _Float16* __restrict__ Khm,
                                                          float* __restrict__ attn) {
  const int lane = threadIdx.x & 31;
  const int wave = threadIdx.x >> 5;
  const int lrow = lane & 15;
  const int lh = lane >> 4;
  const int bh = blockIdx.y;
  const int q0 = blockIdx.x * 16;

  const _Float16* Qb = Qhm + (size_t)bh * SEQ * HEAD_DIM;
  const _Float16* Kb = Khm + (size_t)bh * SEQ * HEAD_DIM;
  float* S = attn + ((size_t)bh * SEQ + q0) * SEQ;

  const _Float16* qr = Qb + (size_t)(q0 + lrow) * HEAD_DIM;
  const v16h a0 = afrag_from_f16(qr, 0, lh);
  const v16h a1 = afrag_from_f16(qr, 32, lh);

  const int key0 = wave * 256;
  for (int t = 0; t < 16; ++t) {
    const int kb = key0 + t * 16;
    const _Float16* kr = Kb + (size_t)(kb + lrow) * HEAD_DIM;
    const v16h b0 = bfrag_from_f16(kr, 0, lh);   // B[k=d][n=key]: K row is contiguous in d
    const v16h b1 = bfrag_from_f16(kr, 32, lh);
    v8f acc = {};
    acc = WMMA_F16(a0, b0, acc);
    acc = WMMA_F16(a1, b1, acc);
#pragma unroll
    for (int r = 0; r < 8; ++r)
      S[(size_t)(lh * 8 + r) * SEQ + (kb + lrow)] = acc[r] * SCALE;
  }
}

// ---------------------------------------------------------------------------
// Fused top-k(64) + softmax, one wave per 2048-element row.  Row is staged
// global -> LDS with CDNA5 async-to-LDS copies, keys held in registers as
// order-preserving u32; 32-step radix select finds the 64th-largest value.
// Dense weights written back in place; (idx, weight) compacted for PV.
// ---------------------------------------------------------------------------
__global__ __launch_bounds__(128) void topk_softmax_kernel(float* __restrict__ attn,
                                                           float* __restrict__ cw,
                                                           unsigned short* __restrict__ ci) {
  __shared__ float srow[4][SEQ];
  __shared__ int scnt[4];
  const int lane = threadIdx.x & 31;
  const int wave = threadIdx.x >> 5;
  const int row = blockIdx.x * 4 + wave;
  float* rowp = attn + (size_t)row * SEQ;

  if (lane == 0) scnt[wave] = 0;

  // Async stage 8KB row into LDS: 16 x b128 per lane, coalesced.
  const uint32_t lds0 = (uint32_t)(uintptr_t)(&srow[wave][0]);
#pragma unroll
  for (int c = 0; c < 16; ++c) {
    const int e = c * 128 + lane * 4;
    const uint32_t lds = lds0 + e * 4;
    const uint64_t g = (uint64_t)(uintptr_t)(rowp + e);
    asm volatile("global_load_async_to_lds_b128 %0, %1, off" ::"v"(lds), "v"(g) : "memory");
  }
  asm volatile("s_wait_asynccnt 0x0" ::: "memory");

  uint32_t u[64];
#pragma unroll
  for (int i = 0; i < 64; ++i) {
    const uint32_t bits = __float_as_uint(srow[wave][lane + i * 32]);
    u[i] = (bits & 0x80000000u) ? ~bits : (bits | 0x80000000u);  // order-preserving key
  }

  // Radix select: threshold = 64th largest key of the row.
  uint32_t prefix = 0;
  int want = TOPK;
  for (int b = 31; b >= 0; --b) {
    const uint32_t hi = (prefix | (1u << b)) >> b;
    int cnt = 0;
#pragma unroll
    for (int i = 0; i < 64; ++i) cnt += ((u[i] >> b) == hi) ? 1 : 0;
#pragma unroll
    for (int off = 16; off >= 1; off >>= 1) cnt += __shfl_xor(cnt, off, 32);
    if (cnt >= want)
      prefix |= (1u << b);
    else
      want -= cnt;
  }
  const uint32_t thr = prefix;

  uint32_t um = 0;
#pragma unroll
  for (int i = 0; i < 64; ++i) um = (u[i] > um) ? u[i] : um;
#pragma unroll
  for (int off = 16; off >= 1; off >>= 1) {
    const uint32_t o = __shfl_xor(um, off, 32);
    um = (o > um) ? o : um;
  }
  const float fmax = ord2f(um);

  float s = 0.f;
#pragma unroll
  for (int i = 0; i < 64; ++i)
    if (u[i] >= thr) s += __expf(ord2f(u[i]) - fmax);
#pragma unroll
  for (int off = 16; off >= 1; off >>= 1) s += __shfl_xor(s, off, 32);
  const float inv = 1.0f / s;

#pragma unroll
  for (int i = 0; i < 64; ++i) {
    const int e = lane + i * 32;
    float w = 0.f;
    if (u[i] >= thr) {
      w = __expf(ord2f(u[i]) - fmax) * inv;
      const int slot = atomicAdd(&scnt[wave], 1);
      if (slot < TOPK) {
        cw[(size_t)row * TOPK + slot] = w;
        ci[(size_t)row * TOPK + slot] = (unsigned short)e;
      }
    }
    rowp[e] = w;  // dense weights: zero outside top-k
  }
}

// ---------------------------------------------------------------------------
// PV with top-k gather (3% density -> 32x fewer FLOPs than dense WMMA PV).
// One wave per query row; lane owns d and d+32; ctx stored f16 row-major for
// the final WMMA projection.
// ---------------------------------------------------------------------------
__global__ __launch_bounds__(256) void pv_gather_kernel(const float* __restrict__ cw,
                                                        const unsigned short* __restrict__ ci,
                                                        const _Float16* __restrict__ Vhm,
                                                        _Float16* __restrict__ ctx) {
  const int lane = threadIdx.x & 31;
  const int wave = threadIdx.x >> 5;
  const int row = blockIdx.x * 8 + wave;  // bh*SEQ + q
  const int bh = row >> 11;
  const int q = row & (SEQ - 1);
  const int b = bh >> 3, h = bh & 7;

  const _Float16* Vb = Vhm + (size_t)bh * SEQ * HEAD_DIM;
  const float* wr = cw + (size_t)row * TOPK;
  const unsigned short* ir = ci + (size_t)row * TOPK;

  float acc0 = 0.f, acc1 = 0.f;
#pragma unroll 4
  for (int p = 0; p < TOPK; ++p) {
    const float w = wr[p];
    const int idx = ir[p];
    const _Float16* vr = Vb + (size_t)idx * HEAD_DIM;
    acc0 += w * (float)vr[lane];
    acc1 += w * (float)vr[lane + 32];
  }
  _Float16* cr = ctx + (size_t)(b * SEQ + q) * D_MODEL + h * HEAD_DIM;
  cr[lane] = (_Float16)acc0;
  cr[lane + 32] = (_Float16)acc1;
}

extern "C" void kernel_launch(void* const* d_in, const int* in_sizes, int n_in, void* d_out,
                              int out_size, void* d_ws, size_t ws_size, hipStream_t stream) {
  const float* query = (const float*)d_in[0];
  const float* key = (const float*)d_in[1];
  const float* value = (const float*)d_in[2];
  const float* wq = (const float*)d_in[3];
  const float* bq = (const float*)d_in[4];
  const float* wk = (const float*)d_in[5];
  const float* bk = (const float*)d_in[6];
  const float* wv = (const float*)d_in[7];
  const float* bv = (const float*)d_in[8];
  const float* wo = (const float*)d_in[9];
  const float* bo = (const float*)d_in[10];
  // d_in[11] = top_k (fixed at 64, baked into TOPK)

  float* out = (float*)d_out;                                   // (B, N, 512) fp32
  float* attn = out + (size_t)BATCH * SEQ * D_MODEL;            // (B, H, N, N) fp32

  // Workspace layout (28 MB total)
  char* ws = (char*)d_ws;
  _Float16* Qhm = (_Float16*)(ws);                              //  4 MB  f16 [B][H][N][64]
  _Float16* Khm = (_Float16*)(ws + (size_t)4 * 1024 * 1024);    //  4 MB
  _Float16* Vhm = (_Float16*)(ws + (size_t)8 * 1024 * 1024);    //  4 MB
  _Float16* ctx = (_Float16*)(ws + (size_t)12 * 1024 * 1024);   //  4 MB  f16 [B*N][512]
  float* cw = (float*)(ws + (size_t)16 * 1024 * 1024);          //  8 MB  compact weights
  unsigned short* ci = (unsigned short*)(ws + (size_t)24 * 1024 * 1024);  // 4 MB indices

  const dim3 gg(D_MODEL / 64, MROWS / 128);  // (8, 32)

  proj_gemm_kernel<float, 0><<<gg, 256, 0, stream>>>(query, wq, bq, Qhm);
  proj_gemm_kernel<float, 0><<<gg, 256, 0, stream>>>(key, wk, bk, Khm);
  proj_gemm_kernel<float, 0><<<gg, 256, 0, stream>>>(value, wv, bv, Vhm);

  attn_scores_kernel<<<dim3(SEQ / 16, BATCH * N_HEADS), 256, 0, stream>>>(Qhm, Khm, attn);

  topk_softmax_kernel<<<(BATCH * N_HEADS * SEQ) / 4, 128, 0, stream>>>(attn, cw, ci);

  pv_gather_kernel<<<(BATCH * N_HEADS * SEQ) / 8, 256, 0, stream>>>(cw, ci, Vhm, ctx);

  proj_gemm_kernel<_Float16, 1><<<gg, 256, 0, stream>>>(ctx, wo, bo, out);
}
